// UltraLSNTBlock_87875030876718
// MI455X (gfx1250) — matrix operand order repaired and checked by reading, not compiled, not run on hardware
//
#include <hip/hip_runtime.h>
#include <math.h>

typedef __bf16 bf16;
typedef __bf16 v16bf __attribute__((ext_vector_type(16)));
typedef float  v8f   __attribute__((ext_vector_type(8)));
typedef float  v4f   __attribute__((ext_vector_type(4)));
typedef unsigned int v4u __attribute__((ext_vector_type(4)));

#define DIM   256
#define NTOK  16384

union FragBF { v16bf v; v4u u[2]; };

// WGP-scope prefetch (scope field = 0 -> pull into all cache levels incl WGP$).
// The clang __builtin_prefetch locality knob only reaches SCOPE_SE/SYS, which
// per the ISA do NOT populate the WGP cache, so use inline asm directly.
__device__ __forceinline__ void prefetch_wgp(const void* p){
  asm volatile("global_prefetch_b8 %0, off" :: "v"(p));
}

// Branchless tanh-approx GELU (~10 VALU ops, no EXEC divergence, no NaN:
// exp->inf gives t -> 1, exp->0 gives t -> -1).
__device__ __forceinline__ float geluf(float x){
  float x3 = x * x * x;
  float y  = 0.7978845608028654f * (x + 0.044715f * x3);
  float e  = __expf(2.f * y);
  float t  = 1.f - 2.f / (e + 1.f);
  return 0.5f * x * (1.f + t);
}
__device__ __forceinline__ float sigmoidf_(float x){ return 1.0f / (1.0f + expf(-x)); }
__device__ __forceinline__ float wredsum(float v){
  v += __shfl_xor(v, 1, 32);
  v += __shfl_xor(v, 2, 32);
  v += __shfl_xor(v, 4, 32);
  v += __shfl_xor(v, 8, 32);
  v += __shfl_xor(v, 16, 32);
  return v;
}

// ---------------------------------------------------------------------------
// Repack f32 row-major [batch, K, N] weights into WMMA B-fragment order:
// frag (kt, nt): lane l holds col = l&15, K = kt*32 + (l>>4)*16 + j, j=0..15,
// stored as 16 contiguous bf16 per lane (two b128 loads in the GEMM).
// ---------------------------------------------------------------------------
__global__ void repack_kernel(const float* __restrict__ src, bf16* __restrict__ dst,
                              int K, int N, int batch){
  long idx = (long)blockIdx.x * blockDim.x + threadIdx.x;
  int lane = idx & 31;
  long fragIdx = idx >> 5;
  int ntiles = N >> 4;
  int ktiles = K >> 5;
  long fragsPerB = (long)ktiles * ntiles;
  long total = (long)batch * fragsPerB;
  if (fragIdx >= total) return;
  int b  = (int)(fragIdx / fragsPerB);
  long f = fragIdx % fragsPerB;
  int kt = (int)(f / ntiles), nt = (int)(f % ntiles);
  int col = lane & 15, ks = (lane >> 4) * 16;
  const float* s = src + (long)b * K * N;
  bf16* d = dst + idx * 16;
#pragma unroll
  for (int j = 0; j < 16; ++j){
    int k = kt * 32 + ks + j;
    d[j] = (bf16)s[(long)k * N + nt * 16 + col];
  }
}

// ---------------------------------------------------------------------------
// LayerNorm (biased var, eps 1e-5): one wave per token row, output bf16.
// ---------------------------------------------------------------------------
__global__ void ln_kernel(const float* __restrict__ x, const float* __restrict__ g,
                          const float* __restrict__ b, bf16* __restrict__ out){
  int w = threadIdx.x >> 5, lane = threadIdx.x & 31;
  long row = (long)blockIdx.x * 8 + w;
  const float* xr = x + row * DIM + lane * 8;
  v4f a0 = *(const v4f*)xr;
  v4f a1 = *(const v4f*)(xr + 4);
  float s = 0.f, sq = 0.f;
#pragma unroll
  for (int i = 0; i < 4; ++i){ s += a0[i] + a1[i]; sq += a0[i]*a0[i] + a1[i]*a1[i]; }
  s = wredsum(s); sq = wredsum(sq);
  float m  = s * (1.f / DIM);
  float var = sq * (1.f / DIM) - m * m;
  float rs = rsqrtf(var + 1e-5f);
  union { bf16 h[8]; v4u u; } o;
  int c0 = lane * 8;
#pragma unroll
  for (int i = 0; i < 4; ++i){
    o.h[i]     = (bf16)((a0[i] - m) * rs * g[c0 + i]     + b[c0 + i]);
    o.h[4 + i] = (bf16)((a1[i] - m) * rs * g[c0 + 4 + i] + b[c0 + 4 + i]);
  }
  *(v4u*)(out + row * DIM + c0) = o.u;
}

// ---------------------------------------------------------------------------
// Uncertainty net: u = softplus(gelu(h@W1+b1)@W2 + b2); block-reduced sum(u).
// ---------------------------------------------------------------------------
__global__ void unc_kernel(const bf16* __restrict__ h, const float* __restrict__ w1,
                           const float* __restrict__ b1, const float* __restrict__ w2,
                           const float* __restrict__ b2, float* __restrict__ u,
                           float* __restrict__ acc){
  __shared__ float bsum;
  int tid = threadIdx.x, w = tid >> 5, lane = tid & 31;
  if (tid == 0) bsum = 0.f;
  __syncthreads();
  long row = (long)blockIdx.x * 8 + w;
  float xv[8];
#pragma unroll
  for (int i = 0; i < 8; ++i) xv[i] = (float)h[row * DIM + lane * 8 + i];
  float sacc[16];
#pragma unroll
  for (int t = 0; t < 16; ++t) sacc[t] = 0.f;
  for (int i = 0; i < 8; ++i){
    const float* wr = w1 + (long)(lane * 8 + i) * 16;
#pragma unroll
    for (int t = 0; t < 16; ++t) sacc[t] += xv[i] * wr[t];
  }
  float uv = 0.f;
  for (int t = 0; t < 16; ++t){
    float sv = wredsum(sacc[t]);
    uv += geluf(sv + b1[t]) * w2[t];
  }
  uv += b2[0];
  float sp = (uv > 20.f) ? uv : log1pf(expf(uv));
  if (lane == 0){ u[row] = sp; atomicAdd(&bsum, sp); }
  __syncthreads();
  if (tid == 0) atomicAdd(&acc[0], bsum);
}

// ---------------------------------------------------------------------------
// Router: logits = [h, u_norm] @ router_w; softmax; top-2; combine weights;
// aux-loss accumulators (usage, sel, logsumexp^2) via LDS pre-reduction.
// ---------------------------------------------------------------------------
__global__ void router_kernel(const bf16* __restrict__ h, const float* __restrict__ u,
                              const float* __restrict__ rw, float* __restrict__ cw,
                              float* __restrict__ acc){
  __shared__ float usageL[8], selL[8], zL;
  int tid = threadIdx.x, w = tid >> 5, lane = tid & 31;
  if (tid < 8){ usageL[tid] = 0.f; selL[tid] = 0.f; }
  if (tid == 0) zL = 0.f;
  __syncthreads();
  long row = (long)blockIdx.x * 8 + w;
  float xv[8];
#pragma unroll
  for (int i = 0; i < 8; ++i) xv[i] = (float)h[row * DIM + lane * 8 + i];
  float lg[8];
#pragma unroll
  for (int e = 0; e < 8; ++e) lg[e] = 0.f;
  for (int i = 0; i < 8; ++i){
    const float* wr = rw + (long)(lane * 8 + i) * 8;
#pragma unroll
    for (int e = 0; e < 8; ++e) lg[e] += xv[i] * wr[e];
  }
  for (int e = 0; e < 8; ++e) lg[e] = wredsum(lg[e]);
  float umean = acc[0] * (1.f / NTOK);
  float un = u[row] / (umean + 1e-8f);
#pragma unroll
  for (int e = 0; e < 8; ++e) lg[e] += un * rw[256 * 8 + e];
  float mx = lg[0];
#pragma unroll
  for (int e = 1; e < 8; ++e) mx = fmaxf(mx, lg[e]);
  float se = 0.f, pr[8];
#pragma unroll
  for (int e = 0; e < 8; ++e){ pr[e] = expf(lg[e] - mx); se += pr[e]; }
  float inv = 1.f / se;
#pragma unroll
  for (int e = 0; e < 8; ++e) pr[e] *= inv;
  int i1 = 0;
#pragma unroll
  for (int e = 1; e < 8; ++e) if (pr[e] > pr[i1]) i1 = e;
  int i2 = (i1 == 0) ? 1 : 0;
#pragma unroll
  for (int e = 0; e < 8; ++e) if (e != i1 && pr[e] > pr[i2]) i2 = e;
  float wsum = pr[i1] + pr[i2];
  float w1n = pr[i1] / wsum, w2n = pr[i2] / wsum;
  if (lane == 0){
    for (int e = 0; e < 8; ++e){
      float c = (e == i1) ? w1n : (e == i2) ? w2n : 0.f;
      cw[row * 8 + e] = c;
      atomicAdd(&usageL[e], pr[e]);
    }
    atomicAdd(&selL[i1], 1.f);
    atomicAdd(&selL[i2], 1.f);
    float lse = mx + logf(se);
    atomicAdd(&zL, lse * lse);
  }
  __syncthreads();
  if (tid < 8){ atomicAdd(&acc[1 + tid], usageL[tid]); atomicAdd(&acc[9 + tid], selL[tid]); }
  if (tid == 0) atomicAdd(&acc[17], zL);
}

__global__ void aux_kernel(const float* __restrict__ acc, float* __restrict__ out){
  float lb = 0.f;
  for (int e = 0; e < 8; ++e){
    float usage = acc[1 + e] * (1.f / NTOK);
    float sel   = acc[9 + e] * (1.f / (NTOK * 2));
    lb += usage * sel;
  }
  lb *= 8.f;
  float z = acc[17] * (1.f / NTOK);
  out[(long)NTOK * DIM] = 0.01f * lb + 0.01f * z;
}

// ---------------------------------------------------------------------------
// Fused FFN: out[n] (+= resid) = sum_e cw[n,e]*(GELU(A W1e + b1e) W2e + b2e).
// Block: 512 thr = 16 waves; tile 128 tokens x 256 out (halves per-WGP L2
// weight traffic vs a 64-token tile); per expert, ff in chunks of 256 through
// a padded LDS bf16 tile. bf16 WMMA, f32 accumulate, 8 output tiles per wave.
// Used for the 8-expert MoE (cw != null, resid = x) and the slow path
// (nExperts = 1, cw = null, resid = null).
// ---------------------------------------------------------------------------
__global__ void __launch_bounds__(512) ffn_kernel(
    const bf16* __restrict__ A, const bf16* __restrict__ w1r, const bf16* __restrict__ w2r,
    const float* __restrict__ b1, const float* __restrict__ b2,
    const float* __restrict__ cw, const float* __restrict__ resid,
    float* __restrict__ out, int nExperts, int FF){
  __shared__ bf16 He[128][264];         // +8 elem pad -> 528B row stride
  __shared__ float cwS[128][8];
  int tid = threadIdx.x, w = tid >> 5, lane = tid & 31;
  int mt = w & 7, half = w >> 3;
  long rowBase = (long)blockIdx.x * 128;
  if (cw){
    int i = tid;
    cwS[i >> 3][i & 7] = cw[rowBase * 8 + i];
    i += 512;
    cwS[i >> 3][i & 7] = cw[rowBase * 8 + i];
  }
  __syncthreads();
  v8f zero = {0.f,0.f,0.f,0.f,0.f,0.f,0.f,0.f};
  v8f Oacc[8];
#pragma unroll
  for (int t = 0; t < 8; ++t) Oacc[t] = zero;

  int nChunks = FF >> 8;       // FF/256
  int NT1 = FF >> 4;           // W1 N-tiles
  int KT2 = FF >> 5;           // W2 K-tiles
  long aRow = rowBase + mt * 16 + (lane & 15);
  int kh = (lane >> 4) * 16;

  for (int e = 0; e < nExperts; ++e){
    for (int c = 0; c < nChunks; ++c){
      v8f Hacc[8];
#pragma unroll
      for (int t = 0; t < 8; ++t) Hacc[t] = zero;
      // ---- phase 1: He_c = GELU(A @ W1[:, chunk] + b1) * cw ----
      for (int kt = 0; kt < 8; ++kt){
        if (kt + 1 < 8){
          const bf16* pf = w1r + (((long)e * 8 * NT1 + (long)(kt + 1) * NT1 +
                                   c * 16 + half * 8) * 32 + lane) * 16;
          prefetch_wgp(pf);   // WGP-scope: stage next k-step's frags in WGP$
        }
        FragBF af;
        const bf16* ap = A + aRow * DIM + kt * 32 + kh;
        af.u[0] = *(const v4u*)ap;
        af.u[1] = *(const v4u*)(ap + 8);
#pragma unroll
        for (int t = 0; t < 8; ++t){
          int ntg = c * 16 + half * 8 + t;
          const bf16* bp = w1r + (((long)e * 8 * NT1 + (long)kt * NT1 + ntg) * 32 + lane) * 16;
          FragBF bfr;
          bfr.u[0] = *(const v4u*)bp;
          bfr.u[1] = *(const v4u*)(bp + 8);
          Hacc[t] = __builtin_amdgcn_wmma_f32_16x16x32_bf16(
              false, af.v, false, bfr.v, (short)0, Hacc[t], false, false);
        }
      }
#pragma unroll
      for (int t = 0; t < 8; ++t){
        int ffc = (half * 8 + t) * 16 + (lane & 15);
        float bias = b1[(long)e * FF + c * 256 + ffc];
#pragma unroll
        for (int r = 0; r < 8; ++r){
          int m = mt * 16 + r + ((lane >> 4) << 3);
          float gl = geluf(Hacc[t][r] + bias);
          if (cw) gl *= cwS[m][e];
          He[m][ffc] = (bf16)gl;
        }
      }
      __syncthreads();
      // ---- phase 2: Oacc += He_c @ W2[chunk, :] ----
      int hRow = mt * 16 + (lane & 15);
      for (int kt = 0; kt < 8; ++kt){
        int ktg = c * 8 + kt;
        if (kt + 1 < 8){
          const bf16* pf = w2r + (((long)e * KT2 * 16 + (long)(ktg + 1) * 16 +
                                   half * 8) * 32 + lane) * 16;
          prefetch_wgp(pf);   // WGP-scope: stage next k-step's frags in WGP$
        }
        FragBF af;
        const bf16* ap = &He[hRow][kt * 32 + kh];
        af.u[0] = *(const v4u*)ap;
        af.u[1] = *(const v4u*)(ap + 8);
#pragma unroll
        for (int t = 0; t < 8; ++t){
          int ntg = half * 8 + t;
          const bf16* bp = w2r + (((long)e * KT2 * 16 + (long)ktg * 16 + ntg) * 32 + lane) * 16;
          FragBF bfr;
          bfr.u[0] = *(const v4u*)bp;
          bfr.u[1] = *(const v4u*)(bp + 8);
          Oacc[t] = __builtin_amdgcn_wmma_f32_16x16x32_bf16(
              false, af.v, false, bfr.v, (short)0, Oacc[t], false, false);
        }
      }
      __syncthreads();
    }
  }
  // ---- epilogue: bias combine + residual ----
#pragma unroll
  for (int t = 0; t < 8; ++t){
    int d = (half * 8 + t) * 16 + (lane & 15);
#pragma unroll
    for (int r = 0; r < 8; ++r){
      int m = mt * 16 + r + ((lane >> 4) << 3);
      long gm = rowBase + m;
      float v = Oacc[t][r];
      if (cw){
        for (int e = 0; e < nExperts; ++e) v += cwS[m][e] * b2[e * DIM + d];
      } else {
        v += b2[d];
      }
      if (resid) v += resid[gm * DIM + d];
      out[gm * DIM + d] = v;
    }
  }
}

// ---------------------------------------------------------------------------
// Single GEMM (fast path): out = A @ W + bias, K = N = 256, bf16 WMMA.
// ---------------------------------------------------------------------------
__global__ void __launch_bounds__(256) gemm_kernel(
    const bf16* __restrict__ A, const bf16* __restrict__ wr,
    const float* __restrict__ bias, float* __restrict__ out){
  int tid = threadIdx.x, w = tid >> 5, lane = tid & 31;
  int mt = w & 3, half = w >> 2;
  long rowBase = (long)blockIdx.x * 64;
  v8f zero = {0.f,0.f,0.f,0.f,0.f,0.f,0.f,0.f};
  v8f Oacc[8];
#pragma unroll
  for (int t = 0; t < 8; ++t) Oacc[t] = zero;
  long aRow = rowBase + mt * 16 + (lane & 15);
  int kh = (lane >> 4) * 16;
  for (int kt = 0; kt < 8; ++kt){
    FragBF af;
    const bf16* ap = A + aRow * DIM + kt * 32 + kh;
    af.u[0] = *(const v4u*)ap;
    af.u[1] = *(const v4u*)(ap + 8);
#pragma unroll
    for (int t = 0; t < 8; ++t){
      int ntg = half * 8 + t;
      const bf16* bp = wr + (((long)kt * 16 + ntg) * 32 + lane) * 16;
      FragBF bfr;
      bfr.u[0] = *(const v4u*)bp;
      bfr.u[1] = *(const v4u*)(bp + 8);
      Oacc[t] = __builtin_amdgcn_wmma_f32_16x16x32_bf16(
          false, af.v, false, bfr.v, (short)0, Oacc[t], false, false);
    }
  }
#pragma unroll
  for (int t = 0; t < 8; ++t){
    int d = (half * 8 + t) * 16 + (lane & 15);
    float bv = bias[d];
#pragma unroll
    for (int r = 0; r < 8; ++r){
      int m = mt * 16 + r + ((lane >> 4) << 3);
      out[(rowBase + m) * DIM + d] = Oacc[t][r] + bv;
    }
  }
}

// ---------------------------------------------------------------------------
// Mode + importance gates: sigmoid(gelu(h@W1+b1)@W2 + b2), 256->64->1.
// ---------------------------------------------------------------------------
__global__ void modeimp_kernel(const bf16* __restrict__ h,
    const float* __restrict__ mw1, const float* __restrict__ mb1,
    const float* __restrict__ mw2, const float* __restrict__ mb2,
    const float* __restrict__ iw1, const float* __restrict__ ib1,
    const float* __restrict__ iw2, const float* __restrict__ ib2,
    float* __restrict__ modebuf, float* __restrict__ impbuf){
  __shared__ float xs[8][DIM];
  int tid = threadIdx.x, w = tid >> 5, lane = tid & 31;
  long row = (long)blockIdx.x * 8 + w;
#pragma unroll
  for (int i = 0; i < 8; ++i) xs[w][lane * 8 + i] = (float)h[row * DIM + lane * 8 + i];
  __syncthreads();
  for (int net = 0; net < 2; ++net){
    const float* w1 = net ? iw1 : mw1;
    const float* b1 = net ? ib1 : mb1;
    const float* w2 = net ? iw2 : mw2;
    const float* b2 = net ? ib2 : mb2;
    float p = 0.f;
    for (int jj = lane; jj < 64; jj += 32){
      float s = 0.f;
      for (int k = 0; k < DIM; ++k) s += xs[w][k] * w1[(long)k * 64 + jj];
      p += geluf(s + b1[jj]) * w2[jj];
    }
    p = wredsum(p);
    if (lane == 0){
      float v = sigmoidf_(p + b2[0]);
      if (net) impbuf[row] = v; else modebuf[row] = v;
    }
  }
}

// ---------------------------------------------------------------------------
// Dual-path epilogue: LN(fast), LN(slow), hard mode select, chg gate over
// concat(h2, out), skip decision, final residual write.
// ---------------------------------------------------------------------------
__global__ void epilogue2_kernel(const float* __restrict__ xmid, const bf16* __restrict__ h2,
    const float* __restrict__ fastlin, const float* __restrict__ slowlin,
    const float* __restrict__ fg, const float* __restrict__ fb,
    const float* __restrict__ sg, const float* __restrict__ sb,
    const float* __restrict__ modebuf, const float* __restrict__ impbuf,
    const float* __restrict__ cw1, const float* __restrict__ cb1,
    const float* __restrict__ cw2, const float* __restrict__ cb2,
    const float* __restrict__ thr_p, float* __restrict__ out){
  __shared__ float cat[8][2 * DIM];
  int tid = threadIdx.x, w = tid >> 5, lane = tid & 31;
  long row = (long)blockIdx.x * 8 + w;
  int c0 = lane * 8;
  float f[8], sl[8];
  float s = 0.f, sq = 0.f;
#pragma unroll
  for (int i = 0; i < 8; ++i){ f[i] = fastlin[row * DIM + c0 + i]; s += f[i]; sq += f[i] * f[i]; }
  s = wredsum(s); sq = wredsum(sq);
  float m = s / DIM, var = sq / DIM - m * m, rs = rsqrtf(var + 1e-5f);
#pragma unroll
  for (int i = 0; i < 8; ++i) f[i] = (f[i] - m) * rs * fg[c0 + i] + fb[c0 + i];
  s = 0.f; sq = 0.f;
#pragma unroll
  for (int i = 0; i < 8; ++i){ sl[i] = slowlin[row * DIM + c0 + i]; s += sl[i]; sq += sl[i] * sl[i]; }
  s = wredsum(s); sq = wredsum(sq);
  m = s / DIM; var = sq / DIM - m * m; rs = rsqrtf(var + 1e-5f);
#pragma unroll
  for (int i = 0; i < 8; ++i) sl[i] = (sl[i] - m) * rs * sg[c0 + i] + sb[c0 + i];
  bool useSlow = modebuf[row] > 0.5f;
  float ov[8], hv[8];
#pragma unroll
  for (int i = 0; i < 8; ++i){
    ov[i] = useSlow ? sl[i] : f[i];
    hv[i] = (float)h2[row * DIM + c0 + i];
    cat[w][c0 + i] = hv[i];
    cat[w][DIM + c0 + i] = ov[i];
  }
  __syncthreads();
  float p = 0.f;
  for (int jj = lane; jj < 64; jj += 32){
    float sacc = 0.f;
    for (int k = 0; k < 2 * DIM; ++k) sacc += cat[w][k] * cw1[(long)k * 64 + jj];
    p += geluf(sacc + cb1[jj]) * cw2[jj];
  }
  p = wredsum(p);
  float chg = sigmoidf_(p + cb2[0]);
  float imp = impbuf[row];
  float score = (1.f - imp) * (1.f - chg);
  float thr = sigmoidf_(thr_p[0]);
  bool skip = score > thr;
#pragma unroll
  for (int i = 0; i < 8; ++i){
    float add = skip ? hv[i] : ov[i];
    out[row * DIM + c0 + i] = xmid[row * DIM + c0 + i] + add;
  }
}

// ---------------------------------------------------------------------------
extern "C" void kernel_launch(void* const* d_in, const int* in_sizes, int n_in,
                              void* d_out, int out_size, void* d_ws, size_t ws_size,
                              hipStream_t stream){
  (void)in_sizes; (void)n_in; (void)out_size; (void)ws_size;
  const float* x        = (const float*)d_in[0];
  const float* ln1_g    = (const float*)d_in[1];
  const float* ln1_b    = (const float*)d_in[2];
  const float* ln2_g    = (const float*)d_in[3];
  const float* ln2_b    = (const float*)d_in[4];
  const float* unc_w1   = (const float*)d_in[5];
  const float* unc_b1   = (const float*)d_in[6];
  const float* unc_w2   = (const float*)d_in[7];
  const float* unc_b2   = (const float*)d_in[8];
  const float* router_w = (const float*)d_in[9];
  const float* exp_w1   = (const float*)d_in[10];
  const float* exp_b1   = (const float*)d_in[11];
  const float* exp_w2   = (const float*)d_in[12];
  const float* exp_b2   = (const float*)d_in[13];
  const float* mode_w1  = (const float*)d_in[14];
  const float* mode_b1  = (const float*)d_in[15];
  const float* mode_w2  = (const float*)d_in[16];
  const float* mode_b2  = (const float*)d_in[17];
  const float* fast_w   = (const float*)d_in[18];
  const float* fast_b   = (const float*)d_in[19];
  const float* fast_ln_g= (const float*)d_in[20];
  const float* fast_ln_b= (const float*)d_in[21];
  const float* slow_w1  = (const float*)d_in[22];
  const float* slow_b1  = (const float*)d_in[23];
  const float* slow_w2  = (const float*)d_in[24];
  const float* slow_b2  = (const float*)d_in[25];
  const float* slow_ln_g= (const float*)d_in[26];
  const float* slow_ln_b= (const float*)d_in[27];
  const float* imp_w1   = (const float*)d_in[28];
  const float* imp_b1   = (const float*)d_in[29];
  const float* imp_w2   = (const float*)d_in[30];
  const float* imp_b2   = (const float*)d_in[31];
  const float* chg_w1   = (const float*)d_in[32];
  const float* chg_b1   = (const float*)d_in[33];
  const float* chg_w2   = (const float*)d_in[34];
  const float* chg_b2   = (const float*)d_in[35];
  const float* skip_thr = (const float*)d_in[36];
  float* out = (float*)d_out;

  char* ws = (char*)d_ws;
  size_t off = 0;
  auto alloc = [&](size_t bytes) -> void* {
    void* p = ws + off;
    off += (bytes + 255) & ~(size_t)255;
    return p;
  };
  bf16*  h1b     = (bf16*) alloc((size_t)NTOK * DIM * 2);
  bf16*  h2b     = (bf16*) alloc((size_t)NTOK * DIM * 2);
  bf16*  w1r     = (bf16*) alloc((size_t)8 * 256 * 1024 * 2);
  bf16*  w2r     = (bf16*) alloc((size_t)8 * 1024 * 256 * 2);
  bf16*  sw1r    = (bf16*) alloc((size_t)256 * 1024 * 2);
  bf16*  sw2r    = (bf16*) alloc((size_t)1024 * 256 * 2);
  bf16*  fwr     = (bf16*) alloc((size_t)256 * 256 * 2);
  float* cw      = (float*)alloc((size_t)NTOK * 8 * 4);
  float* u       = (float*)alloc((size_t)NTOK * 4);
  float* acc     = (float*)alloc(128);
  float* xmid    = (float*)alloc((size_t)NTOK * DIM * 4);
  float* fastlin = (float*)alloc((size_t)NTOK * DIM * 4);
  float* slowlin = (float*)alloc((size_t)NTOK * DIM * 4);
  float* modebuf = (float*)alloc((size_t)NTOK * 4);
  float* impbuf  = (float*)alloc((size_t)NTOK * 4);

  hipMemsetAsync(acc, 0, 128, stream);

  // weight repack into WMMA B-fragment order
  {
    long tot;
    tot = 8L * 8 * 64 * 32;
    repack_kernel<<<(unsigned)((tot + 255) / 256), 256, 0, stream>>>(exp_w1, w1r, 256, 1024, 8);
    tot = 8L * 32 * 16 * 32;
    repack_kernel<<<(unsigned)((tot + 255) / 256), 256, 0, stream>>>(exp_w2, w2r, 1024, 256, 8);
    tot = 1L * 8 * 64 * 32;
    repack_kernel<<<(unsigned)((tot + 255) / 256), 256, 0, stream>>>(slow_w1, sw1r, 256, 1024, 1);
    tot = 1L * 32 * 16 * 32;
    repack_kernel<<<(unsigned)((tot + 255) / 256), 256, 0, stream>>>(slow_w2, sw2r, 1024, 256, 1);
    tot = 1L * 8 * 16 * 32;
    repack_kernel<<<(unsigned)((tot + 255) / 256), 256, 0, stream>>>(fast_w, fwr, 256, 256, 1);
  }

  ln_kernel<<<NTOK / 8, 256, 0, stream>>>(x, ln1_g, ln1_b, h1b);
  unc_kernel<<<NTOK / 8, 256, 0, stream>>>(h1b, unc_w1, unc_b1, unc_w2, unc_b2, u, acc);
  router_kernel<<<NTOK / 8, 256, 0, stream>>>(h1b, u, router_w, cw, acc);
  aux_kernel<<<1, 1, 0, stream>>>(acc, out);

  // MoE: dense 8-expert fused FFN, cw-weighted combine + residual -> xmid
  ffn_kernel<<<NTOK / 128, 512, 0, stream>>>(h1b, w1r, w2r, exp_b1, exp_b2,
                                             cw, x, xmid, 8, 1024);

  ln_kernel<<<NTOK / 8, 256, 0, stream>>>(xmid, ln2_g, ln2_b, h2b);
  modeimp_kernel<<<NTOK / 8, 256, 0, stream>>>(h2b, mode_w1, mode_b1, mode_w2, mode_b2,
                                               imp_w1, imp_b1, imp_w2, imp_b2,
                                               modebuf, impbuf);
  gemm_kernel<<<NTOK / 64, 256, 0, stream>>>(h2b, fwr, fast_b, fastlin);
  ffn_kernel<<<NTOK / 128, 512, 0, stream>>>(h2b, sw1r, sw2r, slow_b1, slow_b2,
                                             nullptr, nullptr, slowlin, 1, 1024);
  epilogue2_kernel<<<NTOK / 8, 256, 0, stream>>>(xmid, h2b, fastlin, slowlin,
                                                 fast_ln_g, fast_ln_b, slow_ln_g, slow_ln_b,
                                                 modebuf, impbuf,
                                                 chg_w1, chg_b1, chg_w2, chg_b2,
                                                 skip_thr, out);
}
